// Conv_6511170421767
// MI455X (gfx1250) — matrix operand order repaired and checked by reading, not compiled, not run on hardware
//
#include <hip/hip_runtime.h>
#include <hip/hip_bf16.h>

// ---------------------------------------------------------------------------
// 3x3 conv (stride 1, pad 1) -> implicit GEMM on CDNA5 WMMA (bf16, f32 acc)
//   M = 256 oc, N = 32*56*56 = 100352 pixels, K = 128*9 = 1152
//
// Fast path (needs ~26.3 MB workspace):
//   ws layout: [zero page 256B][X2: NHWC bf16 25.7MB][W2: [oc][t][c] bf16 0.6MB]
//   conv: no LDS / no barriers. WMMA fragments are contiguous 16B runs in the
//   pre-converted layouts -> pure global_load_b128 + v_wmma hot loop. Padding
//   handled by clamping the per-lane B pointer to the zero page (no aggregate
//   selects -> no scratch spills).
// Fallback path: round-1 LDS-staged kernel (on-the-fly conversion).
// ---------------------------------------------------------------------------

typedef __attribute__((ext_vector_type(16))) __bf16        v16bf;
typedef __attribute__((ext_vector_type(8)))  float         v8f;
typedef __attribute__((ext_vector_type(4)))  unsigned int  v4u;

union Frag {
    v4u   u[2];   // 32 bytes of raw bf16 bits (stays in VGPRs)
    v16bf v;
};

__device__ __forceinline__ unsigned short f32_to_bf16_rne(float f) {
    unsigned int u = __float_as_uint(f);
    u += 0x7FFFu + ((u >> 16) & 1u);   // round-to-nearest-even
    return (unsigned short)(u >> 16);
}

#define C_IN   128
#define K_OC   256
#define HW     3136               // 56*56
#define CHW    (C_IN * HW)
#define KHW    (K_OC * HW)
#define KTOT   1152               // 128*9
#define ZPAD   128                // zero page: 128 bf16 elems = 256 B

// ========================= prep kernels ====================================

// x: NCHW f32 -> NHWC bf16, LDS-tiled 32x32 transpose (pad kills conflicts)
__global__ __launch_bounds__(256)
void prep_x_nhwc_bf16(const float* __restrict__ x, unsigned short* __restrict__ X2)
{
    __shared__ float tile[32][33];
    const int tx  = threadIdx.x & 31;
    const int ty  = threadIdx.x >> 5;         // 0..7
    const int n   = blockIdx.x / 98;          // 98 = 3136/32
    const int hwt = blockIdx.x % 98;
    const int c0  = blockIdx.y * 32;
    const int hw0 = hwt * 32;

    #pragma unroll
    for (int i = 0; i < 4; ++i) {
        const int c = c0 + ty + i * 8;
        tile[ty + i * 8][tx] = x[(size_t)(n * C_IN + c) * HW + hw0 + tx];
    }
    __syncthreads();
    #pragma unroll
    for (int i = 0; i < 4; ++i) {
        const int pix = hw0 + ty + i * 8;
        X2[(size_t)(n * HW + pix) * C_IN + c0 + tx] =
            f32_to_bf16_rne(tile[tx][ty + i * 8]);
    }
}

// w: OIHW f32 ([oc][c][t] flat) -> [oc][t][c] bf16; also zero-fill zero page.
__global__ __launch_bounds__(256)
void prep_w_octc_bf16(const float* __restrict__ wgt,
                      unsigned short* __restrict__ W2,
                      unsigned short* __restrict__ Z)
{
    const int idx = blockIdx.x * 256 + threadIdx.x;   // 294912 total, exact
    if (blockIdx.x == 0 && threadIdx.x < ZPAD) Z[threadIdx.x] = 0;
    const int o = idx / KTOT;
    const int r = idx % KTOT;
    const int t = r / C_IN;
    const int c = r % C_IN;
    W2[idx] = f32_to_bf16_rne(wgt[(size_t)o * KTOT + c * 9 + t]);
}

// ========================= fast main kernel ================================
// Workgroup: 256 thr = 8 waves; tile 128 oc x 128 pix.
// Waves 2(M) x 4(N): each wave 64 oc x 32 pix = 4x2 WMMA tiles.
__global__ __launch_bounds__(256)
void conv3x3_wmma_direct(const unsigned short* __restrict__ Z,     // zero page
                         const unsigned short* __restrict__ X2,
                         const unsigned short* __restrict__ W2,
                         const float* __restrict__ bias,
                         float* __restrict__ out)
{
    const int tid      = threadIdx.x;
    const int pix_base = blockIdx.x * 128;   // 784 blocks, exact
    const int oc_base  = blockIdx.y * 128;   // 2 blocks, exact

    const int lane  = tid & 31;
    const int wid   = tid >> 5;
    const int wm    = (wid & 1) * 64;        // wave oc offset within 128
    const int wn    = (wid >> 1) * 32;       // wave pixel offset within 128
    const int lmod  = lane & 15;             // M-row / N-col within tile
    const int lhalf = lane >> 4;             // K-half selector

    // Per-lane A row pointers: W2[oc][t][c], row stride KTOT elements.
    const unsigned short* ap[4];
    #pragma unroll
    for (int tm = 0; tm < 4; ++tm)
        ap[tm] = W2 + (size_t)(oc_base + wm + tm * 16 + lmod) * KTOT;

    // Per-lane B pixel decomposition (two pixel columns per lane).
    int pn[2], ph[2], pw[2];
    #pragma unroll
    for (int tn = 0; tn < 2; ++tn) {
        const int p = pix_base + wn + tn * 16 + lmod;
        pn[tn] = p / HW;
        const int hw = p % HW;
        ph[tn] = hw / 56;
        pw[tn] = hw % 56;
    }

    v8f acc[4][2];
    #pragma unroll
    for (int tm = 0; tm < 4; ++tm) {
        acc[tm][0] = (v8f){};
        acc[tm][1] = (v8f){};
    }

    for (int t = 0; t < 9; ++t) {
        const int dh = t / 3 - 1;
        const int dw = t % 3 - 1;

        // Per-lane B base pointer for this tap; padding lanes read the zero
        // page (scalar 64-bit pointer cndmask once per tap -- no data selects)
        const unsigned short* bp[2];
        #pragma unroll
        for (int tn = 0; tn < 2; ++tn) {
            const int hx = ph[tn] + dh;
            const int wx = pw[tn] + dw;
            const bool vld = (hx >= 0) & (hx < 56) & (wx >= 0) & (wx < 56);
            const unsigned short* real =
                X2 + ((size_t)(pn[tn] * HW + hx * 56 + wx) * C_IN) + lhalf * 16;
            bp[tn] = vld ? real : (Z + lhalf * 16);
        }

        #pragma unroll
        for (int cb = 0; cb < 4; ++cb) {
            const int c0   = cb * 32;
            const int koff = t * C_IN + c0 + lhalf * 16;  // A k-offset (elems)

            // ---- B fragments: 16 contiguous bf16 per lane -----------------
            // NOTE: zero-page reads at c0 + lhalf*16 stay inside [0,128) elems
            Frag fb[2];
            #pragma unroll
            for (int tn = 0; tn < 2; ++tn) {
                fb[tn].u[0] = *(const v4u*)(bp[tn] + c0);        // k = g*16..
                fb[tn].u[1] = *(const v4u*)(bp[tn] + c0 + 8);    // ..g*16+15
            }

            // ---- A fragments: two 16B k-runs per lane (L2/L0-hot) ---------
            Frag fa[4];
            #pragma unroll
            for (int tm = 0; tm < 4; ++tm) {
                fa[tm].u[0] = *(const v4u*)(ap[tm] + koff);        // k=g*16..
                fa[tm].u[1] = *(const v4u*)(ap[tm] + koff + 8);    // k=g*16+8..
            }

            // ---- 4x2 WMMA ------------------------------------------------
            #pragma unroll
            for (int tm = 0; tm < 4; ++tm)
                #pragma unroll
                for (int tn = 0; tn < 2; ++tn)
                    acc[tm][tn] = __builtin_amdgcn_wmma_f32_16x16x32_bf16(
                        false, fa[tm].v, false, fb[tn].v,
                        (short)0, acc[tm][tn], false, false);
        }
    }

    // ---- epilogue: bias + scatter to NCHW (VGPR v -> M = v + 8*lhalf) -----
    #pragma unroll
    for (int tn = 0; tn < 2; ++tn) {
        const int p  = pix_base + wn + tn * 16 + lmod;
        const int n  = p / HW;
        const int hw = p % HW;
        float* ob = out + (size_t)n * KHW + hw;
        #pragma unroll
        for (int tm = 0; tm < 4; ++tm) {
            #pragma unroll
            for (int v = 0; v < 8; ++v) {
                const int oc = oc_base + wm + tm * 16 + lhalf * 8 + v;
                ob[(size_t)oc * HW] = acc[tm][tn][v] + bias[oc];
            }
        }
    }
}

// ========================= fallback (round-1) ==============================
__global__ __launch_bounds__(256)
void conv3x3_wmma_bf16(const float* __restrict__ x,
                       const float* __restrict__ wgt,
                       const float* __restrict__ bias,
                       float* __restrict__ out)
{
    __shared__ __align__(16) unsigned short sA[64 * 32];
    __shared__ __align__(16) unsigned short sB[128 * 32];

    const int tid      = threadIdx.x;
    const int pix_base = blockIdx.x * 128;
    const int oc_base  = blockIdx.y * 64;

    const int a_row  = tid & 63;
    const int a_cg   = tid >> 6;
    const int b_pix  = tid & 127;
    const int b_half = tid >> 7;

    const int pB  = pix_base + b_pix;
    const int nB  = pB / HW;
    const int hwB = pB % HW;
    const int hB  = hwB / 56, wB = hwB % 56;
    const float* xn = x + (size_t)nB * CHW;

    const int lane  = tid & 31;
    const int wid   = tid >> 5;
    const int wm    = (wid & 1) * 32;
    const int wn    = (wid >> 1) * 32;
    const int lmod  = lane & 15;
    const int lhalf = lane >> 4;

    v8f acc[2][2];
    acc[0][0] = (v8f){}; acc[0][1] = (v8f){};
    acc[1][0] = (v8f){}; acc[1][1] = (v8f){};

    for (int t = 0; t < 9; ++t) {
        const int dh = t / 3 - 1, dw = t % 3 - 1;
        const int hx = hB + dh,   wx = wB + dw;
        const bool vld = (hx >= 0) & (hx < 56) & (wx >= 0) & (wx < 56);
        const float* xtap = xn + hx * 56 + wx;

        for (int cb = 0; cb < 4; ++cb) {
            const int c0 = cb * 32;
            __syncthreads();
            const float* wp = wgt + (size_t)(oc_base + a_row) * KTOT
                                  + (c0 + a_cg * 8) * 9 + t;
            #pragma unroll
            for (int i = 0; i < 8; ++i)
                sA[a_row * 32 + a_cg * 8 + i] = f32_to_bf16_rne(wp[i * 9]);

            const float* xp = xtap + (size_t)(c0 + b_half * 16) * HW;
            #pragma unroll
            for (int i = 0; i < 16; ++i) {
                const float v = vld ? xp[(size_t)i * HW] : 0.0f;
                sB[b_pix * 32 + b_half * 16 + i] = f32_to_bf16_rne(v);
            }
            __syncthreads();

            Frag fa[2], fb[2];
            #pragma unroll
            for (int tm = 0; tm < 2; ++tm) {
                const v4u* pa = (const v4u*)(sA + (wm + tm * 16 + lmod) * 32 + lhalf * 16);
                fa[tm].u[0] = pa[0];
                fa[tm].u[1] = pa[1];
            }
            #pragma unroll
            for (int tn = 0; tn < 2; ++tn) {
                const v4u* pb = (const v4u*)(sB + (wn + tn * 16 + lmod) * 32 + lhalf * 16);
                fb[tn].u[0] = pb[0];
                fb[tn].u[1] = pb[1];
            }
            #pragma unroll
            for (int tm = 0; tm < 2; ++tm)
                #pragma unroll
                for (int tn = 0; tn < 2; ++tn)
                    acc[tm][tn] = __builtin_amdgcn_wmma_f32_16x16x32_bf16(
                        false, fa[tm].v, false, fb[tn].v,
                        (short)0, acc[tm][tn], false, false);
        }
    }

    #pragma unroll
    for (int tn = 0; tn < 2; ++tn) {
        const int p  = pix_base + wn + tn * 16 + lmod;
        const int n  = p / HW;
        const int hw = p % HW;
        float* ob = out + (size_t)n * KHW + hw;
        #pragma unroll
        for (int tm = 0; tm < 2; ++tm)
            #pragma unroll
            for (int v = 0; v < 8; ++v) {
                const int oc = oc_base + wm + tm * 16 + lhalf * 8 + v;
                ob[(size_t)oc * HW] = acc[tm][tn][v] + bias[oc];
            }
    }
}

// ========================= launch ==========================================
extern "C" void kernel_launch(void* const* d_in, const int* in_sizes, int n_in,
                              void* d_out, int out_size, void* d_ws, size_t ws_size,
                              hipStream_t stream) {
    const float* x    = (const float*)d_in[0];
    const float* wgt  = (const float*)d_in[1];
    const float* bias = (const float*)d_in[2];
    float* out = (float*)d_out;

    const size_t X2_elems = (size_t)32 * HW * C_IN;       // 12,845,056
    const size_t W2_elems = (size_t)K_OC * KTOT;          //    294,912
    const size_t need = (ZPAD + X2_elems + W2_elems) * sizeof(unsigned short);

    if (ws_size >= need) {
        unsigned short* Z  = (unsigned short*)d_ws;       // 256 B zero page
        unsigned short* X2 = Z + ZPAD;
        unsigned short* W2 = X2 + X2_elems;
        prep_x_nhwc_bf16<<<dim3(32 * 98, 4), 256, 0, stream>>>(x, X2);
        prep_w_octc_bf16<<<dim3(1152), 256, 0, stream>>>(wgt, W2, Z);
        conv3x3_wmma_direct<<<dim3(784, 2), 256, 0, stream>>>(Z, X2, W2, bias, out);
    } else {
        conv3x3_wmma_bf16<<<dim3(784, 4), 256, 0, stream>>>(x, wgt, bias, out);
    }
}
